// CausalSelfAttention_23854248362738
// MI455X (gfx1250) — matrix operand order repaired and checked by reading, not compiled, not run on hardware
//
#include <hip/hip_runtime.h>
#include <hip/hip_bf16.h>

// ---------------------------------------------------------------------------
// CausalSelfAttention for MI455X (gfx1250): bf16 WMMA everywhere, f32 accum.
// B=4, L=2048, C=1024, H=16, hd=64.
// ---------------------------------------------------------------------------

#define N_EMBD 1024
#define N_HEAD 16
#define HD     64
#define BATCH  4
#define SEQ    2048
#define MROWS  (BATCH * SEQ)   // 8192

typedef __attribute__((ext_vector_type(16))) __bf16 v16bf;
typedef __attribute__((ext_vector_type(8)))  float  v8f;

union Frag {
    v16bf v;
    uint4 u[2];
};

__device__ __forceinline__ unsigned short f2bf(float f) {
    unsigned u = __builtin_bit_cast(unsigned, f);
    unsigned r = u + 0x7FFFu + ((u >> 16) & 1u);   // round-to-nearest-even
    return (unsigned short)(r >> 16);
}

__device__ __forceinline__ v8f v8f_zero() {
    v8f v;
#pragma unroll
    for (int i = 0; i < 8; ++i) v[i] = 0.0f;
    return v;
}

__device__ __forceinline__ v8f wmma_bf16(const v16bf& a, const v16bf& b, const v8f& c) {
    return __builtin_amdgcn_wmma_f32_16x16x32_bf16(
        /*neg_a=*/false, a, /*neg_b=*/false, b,
        /*c_mod=*/(short)0, c, /*reuse_a=*/false, /*reuse_b=*/false);
}

// ---------------------------------------------------------------------------
// f32 -> bf16 conversion (vectorized x4)
// ---------------------------------------------------------------------------
__global__ __launch_bounds__(256) void cvt_f32_bf16(
    const float* __restrict__ src, unsigned short* __restrict__ dst, int n) {
    int i = (blockIdx.x * blockDim.x + threadIdx.x) * 4;
    if (i + 3 < n) {
        float4 f = *(const float4*)(src + i);
        ushort4 o;
        o.x = f2bf(f.x); o.y = f2bf(f.y); o.z = f2bf(f.z); o.w = f2bf(f.w);
        *(ushort4*)(dst + i) = o;
    }
}

// ---------------------------------------------------------------------------
// GEMM: out[m,n] = sum_k A[m,k] * Wt[n,k]   (A: MxK bf16, Wt: NxK bf16)
// 256 threads = 8 waves; wave tile 32x32 (2 M-tiles x 2 N-tiles); block 128x64.
// mode 0: f32 row-major to outf
// mode 1: bf16 to outb at (b,h,l,d)   [Q,K layout]
// mode 2: bf16 to outb at (b,h,d,l)   [V transposed]
// ---------------------------------------------------------------------------
__global__ __launch_bounds__(256) void gemm_bf16(
    const unsigned short* __restrict__ A,
    const unsigned short* __restrict__ Wt,
    unsigned short* __restrict__ outb,
    float* __restrict__ outf,
    int M, int K, int N, int mode) {
    const int lane = threadIdx.x & 31;
    const int w    = threadIdx.x >> 5;
    const int wm   = w & 3;
    const int wn   = w >> 2;
    const int m0   = blockIdx.x * 128 + wm * 32;
    const int n0   = blockIdx.y * 64  + wn * 32;
    const int half = lane >> 4;       // selects A k-chunk / B k-half
    const int l16  = lane & 15;

    v8f acc[2][2];
#pragma unroll
    for (int i = 0; i < 2; ++i)
#pragma unroll
        for (int j = 0; j < 2; ++j) acc[i][j] = v8f_zero();

    const unsigned short* ar[2] = {
        A + (size_t)(m0 + l16) * K,
        A + (size_t)(m0 + 16 + l16) * K };
    const unsigned short* br[2] = {
        Wt + (size_t)(n0 + l16) * K,
        Wt + (size_t)(n0 + 16 + l16) * K };

    for (int k0 = 0; k0 < K; k0 += 32) {
        Frag a[2], b[2];
#pragma unroll
        for (int t = 0; t < 2; ++t) {
            // A fragment 16x32: lane-half selects K chunks {0..7,16..23} / {8..15,24..31}
            a[t].u[0] = *(const uint4*)(ar[t] + k0 + half * 8);
            a[t].u[1] = *(const uint4*)(ar[t] + k0 + 16 + half * 8);
            // B fragment 32x16: lane-half selects K chunks {0..7,16..23} / {8..15,24..31}
            b[t].u[0] = *(const uint4*)(br[t] + k0 + half * 8);
            b[t].u[1] = *(const uint4*)(br[t] + k0 + 16 + half * 8);
        }
#pragma unroll
        for (int mt = 0; mt < 2; ++mt)
#pragma unroll
            for (int nt = 0; nt < 2; ++nt)
                acc[mt][nt] = wmma_bf16(a[mt].v, b[nt].v, acc[mt][nt]);
    }

#pragma unroll
    for (int mt = 0; mt < 2; ++mt) {
#pragma unroll
        for (int nt = 0; nt < 2; ++nt) {
#pragma unroll
            for (int r = 0; r < 8; ++r) {
                const int m = m0 + mt * 16 + r + 8 * half;   // C/D layout row
                const int n = n0 + nt * 16 + l16;            // C/D layout col
                const float v = acc[mt][nt][r];
                if (mode == 0) {
                    outf[(size_t)m * N + n] = v;
                } else if (mode == 1) {   // (b,h,l,d)
                    const int bb = m >> 11, ll = m & 2047, hh = n >> 6, dd = n & 63;
                    outb[(((size_t)(bb * N_HEAD + hh)) * SEQ + ll) * HD + dd] = f2bf(v);
                } else {                  // (b,h,d,l)
                    const int bb = m >> 11, ll = m & 2047, hh = n >> 6, dd = n & 63;
                    outb[(((size_t)(bb * N_HEAD + hh)) * HD + dd) * SEQ + ll] = f2bf(v);
                }
            }
        }
    }
}

// ---------------------------------------------------------------------------
// Flash attention: one block = (b, h, 64-query tile). 128 threads = 4 waves,
// each wave owns 16 query rows. K/V^T blocks of 64 keys staged in LDS.
// Online softmax with wave32 shfl reductions (16-lane halves match C layout).
// Output written bf16 to (B, L, C) for the final projection GEMM.
// ---------------------------------------------------------------------------
__global__ __launch_bounds__(128) void attn_kernel(
    const unsigned short* __restrict__ Q,    // (B,H,L,hd) bf16
    const unsigned short* __restrict__ Kd,   // (B,H,L,hd) bf16
    const unsigned short* __restrict__ Vt,   // (B,H,hd,L) bf16
    const int* __restrict__ amask,           // (B,L) int32
    unsigned short* __restrict__ Y) {        // (B,L,C) bf16
    __shared__ unsigned short Ks[64][72];    // keys x hd
    __shared__ unsigned short Vs[64][72];    // hd x keys (V^T)
    __shared__ unsigned short Ps[4][16][72]; // per-wave P tile 16 x 64

    const int lane = threadIdx.x & 31;
    const int w    = threadIdx.x >> 5;
    const int half = lane >> 4;
    const int l16  = lane & 15;
    const int b    = blockIdx.y >> 4;
    const int h    = blockIdx.y & 15;
    const int q0   = blockIdx.x * 64;
    const int qr   = q0 + w * 16;            // wave's first query row
    const size_t bh = (size_t)(b * N_HEAD + h);

    // Q fragments (two 16x32 A-fragments covering hd=64)
    const unsigned short* qrow = Q + (bh * SEQ + qr + l16) * HD;
    Frag aq[2];
#pragma unroll
    for (int ks = 0; ks < 2; ++ks) {
        aq[ks].u[0] = *(const uint4*)(qrow + ks * 32 + half * 8);
        aq[ks].u[1] = *(const uint4*)(qrow + ks * 32 + 16 + half * 8);
    }

    float mrow[8], lrow[8];
    v8f o[4];
#pragma unroll
    for (int r = 0; r < 8; ++r) { mrow[r] = -1e30f; lrow[r] = 0.0f; }
#pragma unroll
    for (int dt = 0; dt < 4; ++dt) o[dt] = v8f_zero();

    const int njb = blockIdx.x + 1;          // causal: only key blocks <= q tile
    for (int jb = 0; jb < njb; ++jb) {
        const int j0 = jb * 64;

        // Cooperative stage of K (keys x hd) and V^T (hd x keys) into LDS
        {
            const int i  = threadIdx.x >> 1;     // 0..63
            const int hh = threadIdx.x & 1;      // 32-element half
            const unsigned short* kg = Kd + (bh * SEQ + (j0 + i)) * HD + hh * 32;
            const unsigned short* vg = Vt + (bh * HD + i) * SEQ + j0 + hh * 32;
            uint4* ksd = (uint4*)&Ks[i][hh * 32];
            uint4* vsd = (uint4*)&Vs[i][hh * 32];
#pragma unroll
            for (int t = 0; t < 4; ++t) {
                ksd[t] = *(const uint4*)(kg + t * 8);
                vsd[t] = *(const uint4*)(vg + t * 8);
            }
        }
        __syncthreads();

        // S = Q K^T  (16 queries x 64 keys per wave, 4 C-tiles)
        v8f s[4];
#pragma unroll
        for (int nt = 0; nt < 4; ++nt) {
            s[nt] = v8f_zero();
#pragma unroll
            for (int ks = 0; ks < 2; ++ks) {
                Frag bk;
                const unsigned short* kr = &Ks[nt * 16 + l16][ks * 32 + half * 8];
                bk.u[0] = *(const uint4*)(kr);
                bk.u[1] = *(const uint4*)(kr + 16);
                s[nt] = wmma_bf16(aq[ks].v, bk.v, s[nt]);
            }
        }

        // masks + online softmax (C layout: lane%16 = key col, vgpr r = row)
        int keyc[4], mk[4];
#pragma unroll
        for (int nt = 0; nt < 4; ++nt) {
            keyc[nt] = j0 + nt * 16 + l16;
            mk[nt]   = amask[b * SEQ + keyc[nt]];
        }
#pragma unroll
        for (int r = 0; r < 8; ++r) {
            const int q = q0 + w * 16 + r + 8 * half;
            float rm = -1e30f;
#pragma unroll
            for (int nt = 0; nt < 4; ++nt) {
                float sv = s[nt][r] * 0.125f;                 // 1/sqrt(hd)
                if (keyc[nt] > q || mk[nt] == 0) sv = -1e30f; // causal + pad
                s[nt][r] = sv;
                rm = fmaxf(rm, sv);
            }
#pragma unroll
            for (int d = 1; d < 16; d <<= 1)
                rm = fmaxf(rm, __shfl_xor(rm, d, 32));        // stays in 16-lane half
            const float newm = fmaxf(mrow[r], rm);
            float ls = 0.0f;
#pragma unroll
            for (int nt = 0; nt < 4; ++nt) {
                const float p = __expf(s[nt][r] - newm);
                s[nt][r] = p;
                ls += p;
            }
#pragma unroll
            for (int d = 1; d < 16; d <<= 1)
                ls += __shfl_xor(ls, d, 32);
            const float es = __expf(mrow[r] - newm);
            lrow[r] = lrow[r] * es + ls;
            mrow[r] = newm;
#pragma unroll
            for (int dt = 0; dt < 4; ++dt) o[dt][r] *= es;
            // spill P row to LDS (bf16), C layout -> [row][key]
#pragma unroll
            for (int nt = 0; nt < 4; ++nt)
                Ps[w][r + 8 * half][nt * 16 + l16] = f2bf(s[nt][r]);
        }

        // Re-read P as A-fragments (same-wave LDS ops are in order)
        Frag ap[2];
        const unsigned short* pr = &Ps[w][l16][0];
#pragma unroll
        for (int ks = 0; ks < 2; ++ks) {
            ap[ks].u[0] = *(const uint4*)(pr + ks * 32 + half * 8);
            ap[ks].u[1] = *(const uint4*)(pr + ks * 32 + 16 + half * 8);
        }

        // O += P @ V  (B-fragments contiguous thanks to V^T layout in LDS)
#pragma unroll
        for (int dt = 0; dt < 4; ++dt) {
#pragma unroll
            for (int ks = 0; ks < 2; ++ks) {
                Frag bv;
                const unsigned short* vr = &Vs[dt * 16 + l16][ks * 32 + half * 8];
                bv.u[0] = *(const uint4*)(vr);
                bv.u[1] = *(const uint4*)(vr + 16);
                o[dt] = wmma_bf16(ap[ks].v, bv.v, o[dt]);
            }
        }
        __syncthreads();   // protect K/V tiles before next block's staging
    }

    // Normalize and store to (B, L, C) bf16 for the output projection
#pragma unroll
    for (int r = 0; r < 8; ++r) {
        const float inv = lrow[r] > 0.0f ? 1.0f / lrow[r] : 0.0f;
        const int q = q0 + w * 16 + r + 8 * half;
        unsigned short* yr = Y + ((size_t)b * SEQ + q) * N_EMBD + h * HD;
#pragma unroll
        for (int dt = 0; dt < 4; ++dt)
            yr[dt * 16 + l16] = f2bf(o[dt][r] * inv);
    }
}

// ---------------------------------------------------------------------------
// Launch: cvt -> QKV GEMMs -> flash attention -> output projection
// Workspace layout (bf16/ushort), total ~88 MB:
//   xb   @ 0        : 8192*1024
//   wqb  @ 16 MB    : 1024*1024   (x4: wq,wk,wv,wp)
//   qb   @ 24 MB    : B*H*L*hd
//   kb   @ 40 MB
//   vtb  @ 56 MB
//   yb   @ 72 MB
// ---------------------------------------------------------------------------
extern "C" void kernel_launch(void* const* d_in, const int* in_sizes, int n_in,
                              void* d_out, int out_size, void* d_ws, size_t ws_size,
                              hipStream_t stream) {
    const float* x  = (const float*)d_in[0];
    const float* Wq = (const float*)d_in[1];
    const float* Wk = (const float*)d_in[2];
    const float* Wv = (const float*)d_in[3];
    const float* Wp = (const float*)d_in[4];
    const int* am   = (const int*)d_in[5];
    float* out      = (float*)d_out;

    char* ws = (char*)d_ws;
    const size_t SZ_X = (size_t)MROWS * N_EMBD * 2;          // 16 MB
    const size_t SZ_W = (size_t)N_EMBD * N_EMBD * 2;         // 2 MB
    unsigned short* xb  = (unsigned short*)(ws);
    unsigned short* wqb = (unsigned short*)(ws + SZ_X);
    unsigned short* wkb = (unsigned short*)(ws + SZ_X + SZ_W);
    unsigned short* wvb = (unsigned short*)(ws + SZ_X + 2 * SZ_W);
    unsigned short* wpb = (unsigned short*)(ws + SZ_X + 3 * SZ_W);
    unsigned short* qb  = (unsigned short*)(ws + SZ_X + 4 * SZ_W);
    unsigned short* kb  = (unsigned short*)(ws + 2 * SZ_X + 4 * SZ_W);
    unsigned short* vtb = (unsigned short*)(ws + 3 * SZ_X + 4 * SZ_W);
    unsigned short* yb  = (unsigned short*)(ws + 4 * SZ_X + 4 * SZ_W);

    // f32 -> bf16 conversions
    {
        const int nx = MROWS * N_EMBD;                       // 8 Mi elements
        const int nw = N_EMBD * N_EMBD;                      // 1 Mi elements
        cvt_f32_bf16<<<nx / 1024, 256, 0, stream>>>(x,  xb,  nx);
        cvt_f32_bf16<<<nw / 1024, 256, 0, stream>>>(Wq, wqb, nw);
        cvt_f32_bf16<<<nw / 1024, 256, 0, stream>>>(Wk, wkb, nw);
        cvt_f32_bf16<<<nw / 1024, 256, 0, stream>>>(Wv, wvb, nw);
        cvt_f32_bf16<<<nw / 1024, 256, 0, stream>>>(Wp, wpb, nw);
    }

    // QKV projections
    dim3 ggrid(MROWS / 128, N_EMBD / 64);
    gemm_bf16<<<ggrid, 256, 0, stream>>>(xb, wqb, qb,  nullptr, MROWS, N_EMBD, N_EMBD, 1);
    gemm_bf16<<<ggrid, 256, 0, stream>>>(xb, wkb, kb,  nullptr, MROWS, N_EMBD, N_EMBD, 1);
    gemm_bf16<<<ggrid, 256, 0, stream>>>(xb, wvb, vtb, nullptr, MROWS, N_EMBD, N_EMBD, 2);

    // Flash attention: (L/64) x (B*H) blocks, 4 waves each
    dim3 agrid(SEQ / 64, BATCH * N_HEAD);
    attn_kernel<<<agrid, 128, 0, stream>>>(qb, kb, vtb, am, yb);

    // Output projection -> f32 d_out
    gemm_bf16<<<ggrid, 256, 0, stream>>>(yb, wpb, nullptr, out, MROWS, N_EMBD, N_EMBD, 0);

    (void)in_sizes; (void)n_in; (void)out_size; (void)ws_size;
}
